// PhysicsAwareIMUImputer_16844861735028
// MI455X (gfx1250) — matrix operand order repaired and checked by reading, not compiled, not run on hardware
//
#include <hip/hip_runtime.h>
#include <hip/hip_bf16.h>
#include <cmath>

// ---------------------------------------------------------------------------
// PhysicsAwareIMUImputer for MI455X (gfx1250, wave32, WMMA)
//
// Batch-parallel recurrence: 8 workgroups x 16 batch rows; each WG runs the
// full T=1024 loop with LSTM/CfC state resident in LDS. All matmuls are
// v_wmma_f32_16x16x32_bf16 with weights pre-packed into the CDNA5 B-fragment
// lane layout. Weight fragment addresses carry an opaque-zero offset so the
// (t-invariant) loads stay inside the loop as ordinary pipelined b128 loads
// from L2 -- no LICM hoisting -> no spills, no volatile serialization.
// ---------------------------------------------------------------------------

typedef __attribute__((ext_vector_type(16))) __bf16 v16bf;
typedef __attribute__((ext_vector_type(8)))  __bf16 v8bf;
typedef __attribute__((ext_vector_type(8)))  float  v8f;
typedef __attribute__((ext_vector_type(4)))  float  v4f;

#define WMMA_BF16(a, b, c) \
  __builtin_amdgcn_wmma_f32_16x16x32_bf16(false, (a), false, (b), (short)0, (c), false, false)

// ---- problem dims ----
static constexpr int B_  = 128, T_ = 1024, DIN = 13;
static constexpr int NL[3]   = {77, 51, 128};   // CfC layer output width
static constexpr int KIN[3]  = {90, 128, 179};  // CfC layer input width (nin+nh)
static constexpr int NT_[3]  = {5, 4, 8};       // N tiles (ceil(N/16))
static constexpr int KT_[3]  = {3, 4, 6};       // K tiles (ceil(K/32))
static constexpr int NP_[3]  = {80, 64, 128};   // padded N

// ---- workspace layout (bytes). ~0.88 MB total; assumes ws_size >= 1 MB ----
static constexpr size_t WHHP_OFF = 0;                          // bf16 [64 nt][9 kt][512]
static constexpr size_t WHHP_SZ  = (size_t)64 * 9 * 512 * 2;   // 589824
static constexpr size_t LB_OFF   = WHHP_OFF + WHHP_SZ;         // f32 [1024] (bih+bhh)
static constexpr size_t LB_SZ    = 1024 * 4;
static constexpr size_t WPOFF[3] = { LB_OFF + LB_SZ,
                                     LB_OFF + LB_SZ + (size_t)4*5*3*512*2,
                                     LB_OFF + LB_SZ + (size_t)4*5*3*512*2 + (size_t)4*4*4*512*2 };
static constexpr size_t WP_END   = WPOFF[2] + (size_t)4*8*6*512*2;
static constexpr size_t BPOFF[3] = { WP_END, WP_END + 4*80*4, WP_END + 4*80*4 + 4*64*4 };

static constexpr int PREP_TOTAL =
    1024 + 64*9*512 + 4*(5*3 + 4*4 + 8*6)*512 + 4*(80 + 64 + 128);

struct CfcPtrs {
  const float* W[3][4];    // ff1, ff2, ta, tb
  const float* b[3][4];
  const float* mask[3];
};

// ---------------------------------------------------------------------------
// Prep: pack all weights into WMMA B-fragment layout (bf16), fold biases.
// B frag k order used here (matches A loads): lane L -> column n = L%16, K half = 16*(L>=16),
// 16 contiguous bf16 per lane (VGPR v elem e -> K = Khalf + 2v + e).
// LSTM K is extended to 288: [Whh(256) | Wih(13) | zero pad].
// ---------------------------------------------------------------------------
__global__ __launch_bounds__(256)
void imu_prep(const float* __restrict__ Wih, const float* __restrict__ Whh,
              const float* __restrict__ bih, const float* __restrict__ bhh,
              CfcPtrs cp, unsigned char* __restrict__ ws)
{
  int p = blockIdx.x * blockDim.x + threadIdx.x;
  if (p >= PREP_TOTAL) return;

  if (p < 1024) {                                    // fused LSTM bias
    ((float*)(ws + LB_OFF))[p] = bih[p] + bhh[p];
    return;
  }
  p -= 1024;

  if (p < 64*9*512) {                                // packed extended Whh
    int nt = p / (9*512); int r = p % (9*512);
    int kt = r / 512;     int q = r % 512;
    int lane = q >> 4,    j = q & 15;
    int n = nt*16 + (lane & 15);
    int k = kt*32 + ((lane >= 16) ? 16 : 0) + j;
    float v = 0.f;
    if (k < 256)            v = Whh[n*256 + k];
    else if (k < 256 + DIN) v = Wih[n*DIN + (k - 256)];
    ((__bf16*)(ws + WHHP_OFF))[p] = (__bf16)v;
    return;
  }
  p -= 64*9*512;

  for (int l = 0; l < 3; ++l) {                      // packed CfC weights
    int per = NT_[l] * KT_[l] * 512;
    int sz  = 4 * per;
    if (p < sz) {
      int mat = p / per;  int r  = p % per;
      int nt = r / (KT_[l]*512); int r2 = r % (KT_[l]*512);
      int kt = r2 / 512;  int q  = r2 % 512;
      int lane = q >> 4,  j = q & 15;
      int n = nt*16 + (lane & 15);
      int k = kt*32 + ((lane >= 16) ? 16 : 0) + j;
      float v = 0.f;
      if (n < NL[l] && k < KIN[l]) {
        v = cp.W[l][mat][n*KIN[l] + k];
        if (mat < 2) v *= cp.mask[l][n*KIN[l] + k]; // mask only on ff1/ff2
      }
      ((__bf16*)(ws + WPOFF[l]))[p] = (__bf16)v;
      return;
    }
    p -= sz;
  }
  for (int l = 0; l < 3; ++l) {                      // padded CfC biases
    int sz = 4 * NP_[l];
    if (p < sz) {
      int mat = p / NP_[l], n = p % NP_[l];
      ((float*)(ws + BPOFF[l]))[p] = (n < NL[l]) ? cp.b[l][mat][n] : 0.f;
      return;
    }
    p -= sz;
  }
}

// ---------------------------------------------------------------------------
// Main recurrent kernel
// ---------------------------------------------------------------------------
__device__ __forceinline__ float sigm(float v) { return 1.f / (1.f + __expf(-v)); }

// Opaque 0: value LLVM cannot prove constant -> addresses using it are not
// loop-invariant -> weight loads stay in the t-loop (no hoist, no spill),
// while remaining ordinary cached loads (no volatile scope/ordering cost).
__device__ __forceinline__ int opaque_zero() {
  int v = 0;
  asm volatile("" : "+v"(v));
  return v;
}

// A fragment (16x32 bf16) from a row-major bf16 LDS row:
// lane L: M = L%16; K = 16*(L>=16) + 2v+e  -> two 16B ds loads.
__device__ __forceinline__ v16bf load_a_frag(const __bf16* row, int k0, int half8) {
  v8bf c1 = *(const v8bf*)(row + k0 + 2 * half8);
  v8bf c2 = *(const v8bf*)(row + k0 + 2 * half8 + 8);
  return __builtin_shufflevector(c1, c2, 0,1,2,3,4,5,6,7,8,9,10,11,12,13,14,15);
}

// B fragment: 16 contiguous bf16 per lane from the packed weight buffer.
__device__ __forceinline__ v16bf load_b_frag(const __bf16* p) {
  v8bf lo = *(const v8bf*)(p);
  v8bf hi = *(const v8bf*)(p + 8);
  return __builtin_shufflevector(lo, hi, 0,1,2,3,4,5,6,7,8,9,10,11,12,13,14,15);
}

// One CfC layer: wave w owns N-tile w for all four matrices (ff1,ff2,ta,tb),
// so the blend  ff1*(1-t)+t*ff2  is register-local from the 4 accumulators.
template<int NT, int KT, int S, int NVALID, int NPAD, int HOFF, int SOUT, bool LAST>
__device__ __forceinline__ void cfc_layer(const __bf16* zin, __bf16* zout, float* motor,
                                          __bf16* hA, const __bf16* __restrict__ wp,
                                          const float* __restrict__ bp,
                                          int wave, int lane)
{
  if (wave >= NT) return;
  const int half = lane >> 4, col = lane & 15;
  const __bf16* row = zin + col * S;
  const __bf16* wpo = wp + opaque_zero();     // defeat LICM across the t-loop
  v8f a0 = {}, a1 = {}, a2 = {}, a3 = {};
#pragma unroll
  for (int kt = 0; kt < KT; ++kt) {
    v16bf a = load_a_frag(row, kt * 32, half * 8);
    const __bf16* wb = wpo + (size_t)(wave * KT + kt) * 512 + lane * 16;
    v16bf b0 = load_b_frag(wb);
    v16bf b1 = load_b_frag(wb + (size_t)1 * NT * KT * 512);
    v16bf b2 = load_b_frag(wb + (size_t)2 * NT * KT * 512);
    v16bf b3 = load_b_frag(wb + (size_t)3 * NT * KT * 512);
    a0 = WMMA_BF16(a, b0, a0);
    a1 = WMMA_BF16(a, b1, a1);
    a2 = WMMA_BF16(a, b2, a2);
    a3 = WMMA_BF16(a, b3, a3);
  }
  const int n = wave * 16 + col;
  if (n < NVALID) {
    float b1v = bp[0*NPAD + n], b2v = bp[1*NPAD + n];
    float bav = bp[2*NPAD + n], bbv = bp[3*NPAD + n];
#pragma unroll
    for (int r = 0; r < 8; ++r) {
      int m = r + half * 8;                     // C layout: M = r + 8*(lane>=16)
      float ff1 = tanhf(a0[r] + b1v);
      float ff2 = tanhf(a1[r] + b2v);
      float ti  = sigm(a2[r] + bav + a3[r] + bbv);
      float o = ff1 * (1.f - ti) + ti * ff2;
      hA[m * 288 + HOFF + n] = (__bf16)o;       // next-step recurrent h
      if constexpr (LAST) motor[m * 128 + n] = o;
      else                zout[m * SOUT + n] = (__bf16)o;
    }
  }
}

__global__ __launch_bounds__(512, 1)
void imu_main(const float* __restrict__ x, const unsigned char* __restrict__ ws,
              const float* __restrict__ gW, const float* __restrict__ gb,
              const float* __restrict__ aW, const float* __restrict__ ab,
              const float* __restrict__ uW, const float* __restrict__ ub,
              float* __restrict__ out)
{
  __shared__ __attribute__((aligned(32))) __bf16 hA[16 * 288];  // [h(256)|x(13)|pad]
  __shared__ float cS[16 * 256];
  __shared__ float hl[16 * 256];                                // LSTM h output
  __shared__ __attribute__((aligned(32))) __bf16 zA[16 * 192];  // z0 (S=96) / z2 (S=192)
  __shared__ __attribute__((aligned(32))) __bf16 zB[16 * 128];  // z1 (S=128)
  __shared__ float motorS[16 * 128];

  const __bf16* whhP = (const __bf16*)(ws + WHHP_OFF);
  const float*  lb   = (const float*) (ws + LB_OFF);
  const __bf16* wp0  = (const __bf16*)(ws + WPOFF[0]);
  const __bf16* wp1  = (const __bf16*)(ws + WPOFF[1]);
  const __bf16* wp2  = (const __bf16*)(ws + WPOFF[2]);
  const float*  bp0  = (const float*) (ws + BPOFF[0]);
  const float*  bp1  = (const float*) (ws + BPOFF[1]);
  const float*  bp2  = (const float*) (ws + BPOFF[2]);

  const int tid  = threadIdx.x;
  const int wave = tid >> 5, lane = tid & 31;
  const int half = lane >> 4, col = lane & 15;
  const int bgrp = blockIdx.x;                 // 16-row batch group (0..7)

  for (int p = tid; p < 16 * 288; p += 512) hA[p] = (__bf16)0.f;   // h0 = 0, pad = 0
  for (int p = tid; p < 16 * 256; p += 512) cS[p] = 0.f;           // c0 = 0
  __syncthreads();

  for (int t = 0; t < T_; ++t) {
    // stage x_t into the extended-K columns of hA
    if (tid < 16 * DIN) {
      int m = tid / DIN, d = tid % DIN;
      hA[m * 288 + 256 + d] = (__bf16)x[((size_t)(bgrp * 16 + m) * T_ + t) * DIN + d];
    }
    __syncthreads();

    // ---- LSTM: gates = [h|x] @ packed(Whh|Wih), wave w owns hidden block w ----
    {
      const __bf16* row = hA + col * 288;
      const __bf16* whhPo = whhP + opaque_zero();   // defeat LICM across t-loop
      v8f gi = {}, gf = {}, gg = {}, go = {};
#pragma unroll
      for (int kt = 0; kt < 9; ++kt) {
        v16bf a = load_a_frag(row, kt * 32, half * 8);
        const __bf16* base = whhPo + (size_t)kt * 512 + lane * 16;
        v16bf bi  = load_b_frag(base + (size_t)((0 * 16 + wave) * 9) * 512);
        v16bf bf  = load_b_frag(base + (size_t)((1 * 16 + wave) * 9) * 512);
        v16bf bgv = load_b_frag(base + (size_t)((2 * 16 + wave) * 9) * 512);
        v16bf bo  = load_b_frag(base + (size_t)((3 * 16 + wave) * 9) * 512);
        gi = WMMA_BF16(a, bi,  gi);
        gf = WMMA_BF16(a, bf,  gf);
        gg = WMMA_BF16(a, bgv, gg);
        go = WMMA_BF16(a, bo,  go);
      }
      int j = wave * 16 + col;
      float bi_ = lb[j], bf_ = lb[j + 256], bg_ = lb[j + 512], bo_ = lb[j + 768];
#pragma unroll
      for (int r = 0; r < 8; ++r) {
        int m = r + half * 8;
        float cv = sigm(gf[r] + bf_) * cS[m * 256 + j] + sigm(gi[r] + bi_) * tanhf(gg[r] + bg_);
        cS[m * 256 + j] = cv;
        hl[m * 256 + j] = sigm(go[r] + bo_) * tanhf(cv);
      }
    }
    __syncthreads();

    // ---- build z0 = [x|hl[:, :77]|pad] (zA, S=96) and h-part of z1 (zB) ----
    for (int p = tid; p < 16 * 96; p += 512) {
      int m = p / 96, k = p % 96;
      __bf16 v;
      if (k < DIN)     v = hA[m * 288 + 256 + k];
      else if (k < 90) v = (__bf16)hl[m * 256 + (k - DIN)];
      else             v = (__bf16)0.f;
      zA[p] = v;
    }
    for (int p = tid; p < 16 * 51; p += 512) {
      int m = p / 51, k = p % 51;
      zB[m * 128 + 77 + k] = (__bf16)hl[m * 256 + 77 + k];
    }
    __syncthreads();

    // ---- CfC layer 0: z0(90) -> 77; writes zB[:, :77] and hA[:, 0:77] ----
    cfc_layer<5, 3, 96, 77, 80, 0, 128, false>(zA, zB, nullptr, hA, wp0, bp0, wave, lane);
    __syncthreads();

    // ---- CfC layer 1 (waves 0..3) + fill z2 h-part into zA (waves 4..15) ----
    cfc_layer<4, 4, 128, 51, 64, 77, 192, false>(zB, zA, nullptr, hA, wp1, bp1, wave, lane);
    if (wave >= 4) {
      for (int p = tid - 128; p < 16 * 141; p += 384) {
        int m = p / 141, k = 51 + p % 141;     // z2[51..178] = hl[:,128:256]
        zA[m * 192 + k] = (k < 179) ? (__bf16)hl[m * 256 + (k + 77)] : (__bf16)0.f;
      }
    }
    __syncthreads();

    // ---- CfC layer 2: z2(179) -> 128 (motor); writes hA[:, 128:256] ----
    cfc_layer<8, 6, 192, 128, 128, 128, 1, true>(zA, nullptr, motorS, hA, wp2, bp2, wave, lane);
    __syncthreads();

    // ---- output heads: 16 rows x 12 outputs ----
    if (tid < 192) {
      int m = tid / 12, q = tid % 12;
      const float* mr = motorS + m * 128;
      const float* W; float bv;
      if (q < 3)      { W = gW + q * 128;       bv = gb[q]; }
      else if (q < 6) { W = aW + (q - 3) * 128; bv = ab[q - 3]; }
      else            { W = uW + (q - 6) * 128; bv = ub[q - 6]; }
      W += opaque_zero();                       // defeat LICM across t-loop
      float s = bv;
#pragma unroll 4
      for (int k = 0; k < 32; ++k) {
        v4f wv = *(const v4f*)(W + 4 * k);      // plain b128 loads, stay in-loop
        v4f mv = *(const v4f*)(mr + 4 * k);
        s += wv.x * mv.x + wv.y * mv.y + wv.z * mv.z + wv.w * mv.w;
      }
      size_t base = ((size_t)(bgrp * 16 + m) * T_ + t) * 6;
      if (q < 6) out[base + q] = s;                                 // pred
      else       out[(size_t)B_ * T_ * 6 + base + (q - 6)] =        // softplus
                   (s > 20.f) ? s : log1pf(__expf(s));
    }
    __syncthreads();
  }
}

// ---------------------------------------------------------------------------
extern "C" void kernel_launch(void* const* d_in, const int* in_sizes, int n_in,
                              void* d_out, int out_size, void* d_ws, size_t ws_size,
                              hipStream_t stream) {
  (void)in_sizes; (void)n_in; (void)out_size; (void)ws_size;

  const float* x    = (const float*)d_in[0];
  const float* Wih  = (const float*)d_in[1];
  const float* Whh  = (const float*)d_in[2];
  const float* bih  = (const float*)d_in[3];
  const float* bhh  = (const float*)d_in[4];

  CfcPtrs cp;
  for (int l = 0; l < 3; ++l) {
    int base = 5 + l * 9;   // Wff1,bff1,Wff2,bff2,Wta,bta,Wtb,btb,mask
    cp.W[l][0] = (const float*)d_in[base + 0]; cp.b[l][0] = (const float*)d_in[base + 1];
    cp.W[l][1] = (const float*)d_in[base + 2]; cp.b[l][1] = (const float*)d_in[base + 3];
    cp.W[l][2] = (const float*)d_in[base + 4]; cp.b[l][2] = (const float*)d_in[base + 5];
    cp.W[l][3] = (const float*)d_in[base + 6]; cp.b[l][3] = (const float*)d_in[base + 7];
    cp.mask[l] = (const float*)d_in[base + 8];
  }
  const float* gW = (const float*)d_in[32]; const float* gb = (const float*)d_in[33];
  const float* aW = (const float*)d_in[34]; const float* ab = (const float*)d_in[35];
  const float* uW = (const float*)d_in[36]; const float* ub = (const float*)d_in[37];

  unsigned char* ws = (unsigned char*)d_ws;

  imu_prep<<<(PREP_TOTAL + 255) / 256, 256, 0, stream>>>(Wih, Whh, bih, bhh, cp, ws);
  imu_main<<<8, 512, 0, stream>>>(x, ws, gW, gb, aW, ab, uW, ub, (float*)d_out);
}
